// IPEXTransformerQLinear_83116207112270
// MI455X (gfx1250) — matrix operand order repaired and checked by reading, not compiled, not run on hardware
//
#include <hip/hip_runtime.h>

typedef unsigned short u16;
typedef unsigned int   u32;

typedef __attribute__((ext_vector_type(16))) _Float16 v16h;
typedef __attribute__((ext_vector_type(2)))  _Float16 h2;
typedef __attribute__((ext_vector_type(8)))  float    v8f;

#define MROWS 4096          // B*S
#define KDIM  4096
#define NDIM  11008
#define GSZ   128           // quant group size along K
#define BM    128
#define BN    128
#define BK    64            // K staged per LDS chunk (2 chunks per quant group)
#define LDSTRIDE 72         // padded row length in f16 elems (64 + 8), 144B = 9*16B aligned
#define NTHREADS 256

union Frag {
  uint4 q[2];
  v16h  v;
};

__global__ __launch_bounds__(NTHREADS)
void q4gemm_wmma_f16(const float* __restrict__ X, const int* __restrict__ W,
                     const float* __restrict__ Sc, const int* __restrict__ Zp,
                     float* __restrict__ O) {
  __shared__ u16 sA[BM * LDSTRIDE];   // [m][k] f16
  __shared__ u16 sB[BN * LDSTRIDE];   // [n][k] f16 (B operand wants K contiguous per column)

  const int t      = threadIdx.x;
  const int lane   = t & 31;
  const int wid    = t >> 5;
  const int wave_m = wid >> 2;              // 0..1  -> 64 rows each
  const int wave_n = wid & 3;               // 0..3  -> 32 cols each
  const int m0 = blockIdx.y * BM;
  const int n0 = blockIdx.x * BN;

  const v8f zero = {0.f, 0.f, 0.f, 0.f, 0.f, 0.f, 0.f, 0.f};
  v8f acc[4][2];
#pragma unroll
  for (int i = 0; i < 4; ++i)
#pragma unroll
    for (int j = 0; j < 2; ++j) acc[i][j] = zero;

  // staging decomposition (invariant per thread)
  const int arow0 = t >> 4;            // 0..15, A tile row group
  const int acol4 = (t & 15) * 4;      // float4 column within 64-wide K chunk
  const int wrow0 = t >> 5;            // 0..7, packed-W row group
  const int wcol  = t & 31;            // base n offset (n = wcol + 32*j)

  // packed per-column dequant constants, refreshed once per quant group
  h2 smul[4];   // {s, s}
  h2 zadd[4];   // {-(1024+zp), -(1024+zp)}  (exact in f16)

  for (int kc = 0; kc < KDIM / BK; ++kc) {
    const int k0  = kc * BK;
    const int kh0 = k0 >> 1;

    if ((kc & 1) == 0) {
      const int g = k0 / GSZ;
#pragma unroll
      for (int j = 0; j < 4; ++j) {
        const int n = n0 + wcol + 32 * j;
        const float s  = Sc[(size_t)g * NDIM + n];
        const float zn = -(1024.0f + (float)Zp[(size_t)g * NDIM + n]);
        const _Float16 sh = (_Float16)s;
        const _Float16 zh = (_Float16)zn;
        smul[j] = (h2){sh, sh};
        zadd[j] = (h2){zh, zh};
      }
    }

    // ---- stage A: fp32 -> packed f16 into LDS [m][k] (v_cvt_pk_rtz) ----
#pragma unroll
    for (int i = 0; i < 8; ++i) {
      const int row = arow0 + 16 * i;
      const float4 a = *(const float4*)(X + (size_t)(m0 + row) * KDIM + k0 + acol4);
      uint2 pk;
      pk.x = __builtin_bit_cast(u32, __builtin_amdgcn_cvt_pkrtz(a.x, a.y));
      pk.y = __builtin_bit_cast(u32, __builtin_amdgcn_cvt_pkrtz(a.z, a.w));
      *(uint2*)&sA[row * LDSTRIDE + acol4] = pk;
    }

    // ---- stage B: nibble -> f16 via magic 0x6400, packed dequant, LDS [n][k] ----
#pragma unroll
    for (int ir = 0; ir < 4; ++ir) {
      const int r = wrow0 + 8 * ir;          // packed row (kh) within chunk, 0..31
#pragma unroll
      for (int j = 0; j < 4; ++j) {
        const int n = wcol + 32 * j;
        const u32 w = (u32)W[(size_t)(kh0 + r) * NDIM + n0 + n];
        // lo nibble -> lane 0 (even k), hi nibble -> lane 1 (odd k); each = 1024 + q
        const u32 bits = 0x64006400u | (w & 0xFu) | ((w & 0xF0u) << 12);
        const h2 q2 = __builtin_bit_cast(h2, bits);
        const h2 d  = (q2 + zadd[j]) * smul[j];   // v_pk_add_f16 + v_pk_mul_f16
        *(u32*)&sB[n * LDSTRIDE + 2 * r] = __builtin_bit_cast(u32, d);
      }
    }

    __syncthreads();

    // ---- prefetch next K chunk while we compute this one ----
    if (kc + 1 < KDIM / BK) {
      __builtin_prefetch(X + (size_t)(m0 + arow0) * KDIM + k0 + BK + acol4, 0, 1);
      __builtin_prefetch(W + (size_t)(kh0 + 32 + wrow0) * NDIM + n0 + wcol, 0, 1);
    }

    // ---- compute: two K=32 WMMA steps per chunk ----
#pragma unroll
    for (int kk = 0; kk < 2; ++kk) {
      Frag a[4], b[2];
      // A fragment: lanes 0-15 take K [0..7],[16..23]; lanes 16-31 take K [8..15],[24..31]
      const int klo = kk * 32 + ((lane >> 4) ? 8 : 0);
#pragma unroll
      for (int i = 0; i < 4; ++i) {
        const int row = wave_m * 64 + i * 16 + (lane & 15);
        a[i].q[0] = *(const uint4*)&sA[row * LDSTRIDE + klo];
        a[i].q[1] = *(const uint4*)&sA[row * LDSTRIDE + klo + 16];
      }
      // B fragment: lane = column; lanes 0-15 hold K [0..7],[16..23]; lanes 16-31 hold K [8..15],[24..31]
      const int kbo = kk * 32 + ((lane >> 4) ? 8 : 0);
#pragma unroll
      for (int j = 0; j < 2; ++j) {
        const int col = wave_n * 32 + j * 16 + (lane & 15);
        const uint4* p = (const uint4*)&sB[col * LDSTRIDE + kbo];
        b[j].q[0] = p[0];
        b[j].q[1] = p[2];
      }
#pragma unroll
      for (int i = 0; i < 4; ++i)
#pragma unroll
        for (int j = 0; j < 2; ++j)
          acc[i][j] = __builtin_amdgcn_wmma_f32_16x16x32_f16(
              false, a[i].v, false, b[j].v, (short)0, acc[i][j], false, false);
    }

    __syncthreads();
  }

  // ---- epilogue: C/D layout = 8 VGPRs, lanes 0-15 rows M=v, lanes 16-31 rows M=v+8 ----
#pragma unroll
  for (int i = 0; i < 4; ++i) {
    const int row = m0 + wave_m * 64 + i * 16 + ((lane >> 4) * 8);
#pragma unroll
    for (int j = 0; j < 2; ++j) {
      const int col = n0 + wave_n * 32 + j * 16 + (lane & 15);
      float* dst = O + (size_t)row * NDIM + col;
#pragma unroll
      for (int v = 0; v < 8; ++v)
        dst[(size_t)v * NDIM] = acc[i][j][v];
    }
  }
}

extern "C" void kernel_launch(void* const* d_in, const int* in_sizes, int n_in,
                              void* d_out, int out_size, void* d_ws, size_t ws_size,
                              hipStream_t stream) {
  (void)in_sizes; (void)n_in; (void)out_size; (void)d_ws; (void)ws_size;
  const float* X  = (const float*)d_in[0];
  const int*   W  = (const int*)d_in[1];
  const float* Sc = (const float*)d_in[2];
  const int*   Zp = (const int*)d_in[3];
  float*       O  = (float*)d_out;

  dim3 grid(NDIM / BN, MROWS / BM);   // 86 x 32
  dim3 block(NTHREADS);
  q4gemm_wmma_f16<<<grid, block, 0, stream>>>(X, W, Sc, Zp, O);
}
